// LSTMFeatureExtractor_39496519254136
// MI455X (gfx1250) — matrix operand order repaired and checked by reading, not compiled, not run on hardware
//
#include <hip/hip_runtime.h>

typedef __bf16 bf16_t;
typedef bf16_t bf16x16 __attribute__((ext_vector_type(16)));
typedef bf16_t bf16x4  __attribute__((ext_vector_type(4)));
typedef float  f32x8   __attribute__((ext_vector_type(8)));
typedef float  f32x4   __attribute__((ext_vector_type(4)));

constexpr int Bsz = 256;   // batch
constexpr int Tsz = 512;   // seq len
constexpr int Din = 256;   // input dim (== 2H for layer 1)
constexpr int H   = 128;   // hidden
constexpr int G   = 512;   // 4H gates
constexpr int BT  = Bsz * Tsz;

// ---------------------------------------------------------------------------
// fp32 -> bf16 conversion (4 elements / thread)
// ---------------------------------------------------------------------------
__global__ __launch_bounds__(256) void cvt_f32_bf16(const float* __restrict__ src,
                                                    bf16_t* __restrict__ dst, int n) {
  int i = (blockIdx.x * 256 + threadIdx.x) * 4;
  if (i + 4 <= n) {
    f32x4 v = *(const f32x4*)(src + i);
    bf16x4 o;
    o.x = (bf16_t)v.x; o.y = (bf16_t)v.y; o.z = (bf16_t)v.z; o.w = (bf16_t)v.w;
    *(bf16x4*)(dst + i) = o;
  } else {
    for (; i < n; ++i) dst[i] = (bf16_t)src[i];
  }
}

// ---------------------------------------------------------------------------
// WMMA fragment loaders (ISA 7.12.2 layouts, K contiguous in memory)
// A: 16x32 (MxK) bf16.  lanes 0-15: row M=lane, K in {kb..kb+7, kb+16..kb+23},
//                       kb = 0 (lanes<16) / 8 (lanes>=16), 2 elems per VGPR.
// ---------------------------------------------------------------------------
__device__ __forceinline__ bf16x16 load_a_frag(const bf16_t* base, int ld, int lane) {
  const int r  = lane & 15;
  const int kb = (lane >> 4) << 3;          // 0 or 8
  const bf16_t* p = base + r * ld + kb;
  bf16x16 f;
  *(f32x4*)&f       = *(const f32x4*)(p);       // K = kb .. kb+7
  *((f32x4*)&f + 1) = *(const f32x4*)(p + 16);  // K = kb+16 .. kb+23
  return f;
}

// B: 32x16 (KxN), stored as W row-major [N, K] (i.e. B = W^T, K contiguous).
// lanes 0-15: col N=lane, K in {0..7, 16..23} ; lanes 16-31: col N=lane-16, K in {8..15, 24..31}.
__device__ __forceinline__ bf16x16 load_b_frag(const bf16_t* base, int ld, int lane) {
  const int n  = lane & 15;
  const int kb = (lane >> 4) << 3;          // 0 or 8
  const bf16_t* p = base + n * ld + kb;
  bf16x16 f;
  *(f32x4*)&f       = *(const f32x4*)(p);       // K = kb .. kb+7
  *((f32x4*)&f + 1) = *(const f32x4*)(p + 16);  // K = kb+16 .. kb+23
  return f;
}

// ---------------------------------------------------------------------------
// CDNA5 async copy: global -> LDS, 16B per lane, tracked by ASYNCcnt.
// (inline asm: portable across ROCm-7.2 / amdgpu-toolchain builtin arities)
// ---------------------------------------------------------------------------
__device__ __forceinline__ void async_copy_b128(void* lds_ptr, const void* gptr) {
  // Low 32 bits of the flat shared address == LDS byte address (aperture
  // stripped); VDST holds the LDS address, VADDR the 64-bit global address.
  unsigned int loff = (unsigned int)(unsigned long long)lds_ptr;
  asm volatile("global_load_async_to_lds_b128 %0, %1, off"
               :: "v"(loff), "v"(gptr) : "memory");
}
__device__ __forceinline__ void wait_asynccnt0() {
  asm volatile("s_wait_asynccnt 0" ::: "memory");
}

// ---------------------------------------------------------------------------
// Input projection GEMM:  XP[dir][t][b][g] = sum_k X[b*T+t][k] * W[dir][g][k]
//                                            + b_ih[g] + b_hh[g]
// Block: 256 thr (8 waves as 2x4), tile 64 rows x 128 cols, K streamed by 32.
// Output is time-major fp32 so the scan consumes one contiguous 32KB slab/step.
// ---------------------------------------------------------------------------
__global__ __launch_bounds__(256) void xproj_gemm(
    const bf16_t* __restrict__ X,                               // [BT, 256]
    const bf16_t* __restrict__ Wf, const bf16_t* __restrict__ Wb,  // [512, 256]
    const float* __restrict__ bihf, const float* __restrict__ bhhf,
    const float* __restrict__ bihb, const float* __restrict__ bhhb,
    float* __restrict__ XPf, float* __restrict__ XPb)           // [T, B, 512]
{
  const int lane = threadIdx.x & 31;
  const int wave = threadIdx.x >> 5;
  const int wrow = wave >> 2;               // 0..1
  const int wcol = wave & 3;                // 0..3
  const int dir  = blockIdx.z;

  const bf16_t* W  = dir ? Wb  : Wf;
  const float* bih = dir ? bihb : bihf;
  const float* bhh = dir ? bhhb : bhhf;
  float* XP        = dir ? XPb : XPf;

  const int m0 = blockIdx.x * 64 + wrow * 32;    // two 16-row tiles
  const int n0 = blockIdx.y * 128 + wcol * 32;   // two 16-col tiles

  f32x8 acc[2][2];
#pragma unroll
  for (int mi = 0; mi < 2; ++mi)
#pragma unroll
    for (int ni = 0; ni < 2; ++ni)
#pragma unroll
      for (int r = 0; r < 8; ++r) acc[mi][ni][r] = 0.0f;

  for (int kc = 0; kc < Din; kc += 32) {
    bf16x16 a0 = load_a_frag(X + (size_t)m0 * Din + kc, Din, lane);
    bf16x16 a1 = load_a_frag(X + (size_t)(m0 + 16) * Din + kc, Din, lane);
    bf16x16 b0 = load_b_frag(W + (size_t)n0 * Din + kc, Din, lane);
    bf16x16 b1 = load_b_frag(W + (size_t)(n0 + 16) * Din + kc, Din, lane);
    acc[0][0] = __builtin_amdgcn_wmma_f32_16x16x32_bf16(false, a0, false, b0, (short)0, acc[0][0], false, false);
    acc[0][1] = __builtin_amdgcn_wmma_f32_16x16x32_bf16(false, a0, false, b1, (short)0, acc[0][1], false, false);
    acc[1][0] = __builtin_amdgcn_wmma_f32_16x16x32_bf16(false, a1, false, b0, (short)0, acc[1][0], false, false);
    acc[1][1] = __builtin_amdgcn_wmma_f32_16x16x32_bf16(false, a1, false, b1, (short)0, acc[1][1], false, false);
  }

  // C/D layout: VGPR r -> M = r (+8 for upper lanes), lane&15 -> N
  const int nl = lane & 15;
  const int mh = (lane >> 4) * 8;
#pragma unroll
  for (int mi = 0; mi < 2; ++mi) {
#pragma unroll
    for (int ni = 0; ni < 2; ++ni) {
      const int col  = n0 + ni * 16 + nl;
      const float bias = bih[col] + bhh[col];
#pragma unroll
      for (int r = 0; r < 8; ++r) {
        const int m = m0 + mi * 16 + mh + r;   // m = b*T + t
        const int b = m >> 9;                  // / 512
        const int t = m & 511;
        XP[((size_t)t * Bsz + b) * G + col] = acc[mi][ni][r] + bias;
      }
    }
  }
}

// ---------------------------------------------------------------------------
// Recurrent scan. One WG = 16 batch rows x one direction, 512 thr (16 waves).
// LDS: w_hh (bf16 128KB) + 2 x slab (fp32 32KB, async double buffer)
//      + gates (fp32 32KB) + c (fp32 8KB) + h (bf16 4KB) = 236KB.
// Per step: async-copy next slab, WMMA against LDS w_hh (2 N-tiles x 4 K per
// wave), gate activations + cell update, h -> LDS (+ layer-1 input / output).
// ---------------------------------------------------------------------------
#define OFF_SLAB0 131072
#define OFF_GATES (131072 + 65536)
#define OFF_C     (131072 + 65536 + 32768)
#define OFF_H     (131072 + 65536 + 32768 + 8192)
#define SCAN_SMEM (131072 + 65536 + 32768 + 8192 + 4096)

__global__ __launch_bounds__(512) void lstm_scan(
    const float* __restrict__ XPf, const float* __restrict__ XPb,   // [T, B, 512]
    const bf16_t* __restrict__ Whhf, const bf16_t* __restrict__ Whhb, // [512, 128]
    bf16_t* __restrict__ Xout,     // layer0: [B, T, 256] bf16 (null for layer1)
    float* __restrict__ HFinal)    // layer1: [B, 256] fp32  (null for layer0)
{
  extern __shared__ char smem[];
  bf16_t* sW   = (bf16_t*)(smem);                 // [512][128]
  float*  sGat = (float*)(smem + OFF_GATES);      // [16][512]
  float*  sC   = (float*)(smem + OFF_C);          // [16][128]
  bf16_t* sH   = (bf16_t*)(smem + OFF_H);         // [16][128]
  // Slab double buffer lives at OFF_SLAB0 + {0, 32768}; pointers are computed
  // per-iteration from `cur` (no LDS-pointer arrays: lld can't relocate them).

  const int tid  = threadIdx.x;
  const int lane = tid & 31;
  const int wave = tid >> 5;                      // 0..15
  const int dir  = blockIdx.y;
  const int b0   = blockIdx.x * 16;

  const float*  XP  = dir ? XPb  : XPf;
  const bf16_t* Whh = dir ? Whhb : Whhf;

  // Stage w_hh into LDS (8192 x 16B), zero cell/hidden state.
  {
    const f32x4* src = (const f32x4*)Whh;
    f32x4* dst = (f32x4*)sW;
    for (int i = tid; i < 8192; i += 512) dst[i] = src[i];
  }
  for (int i = tid; i < 2048; i += 512) sC[i] = 0.0f;
  for (int i = tid; i < 2048; i += 512) sH[i] = (bf16_t)0.0f;

  // Kick off the async copy of the first slab (32KB = 512 thr x 4 x 16B).
  {
    const int t0 = dir ? (Tsz - 1) : 0;
    const char* g = (const char*)(XP + ((size_t)t0 * Bsz + b0) * G);
    char* l = smem + OFF_SLAB0;
#pragma unroll
    for (int i = 0; i < 4; ++i)
      async_copy_b128(l + tid * 16 + i * 8192, g + tid * 16 + i * 8192);
  }
  wait_asynccnt0();
  __syncthreads();

  const int nl = lane & 15;
  const int mh = (lane >> 4) * 8;

  int cur = 0;
  for (int step = 0; step < Tsz; ++step) {
    const int t = dir ? (Tsz - 1 - step) : step;

    // Issue async copy of the NEXT slab into the alternate buffer; it lands
    // while this step's WMMA + cell update run (ASYNCcnt overlap).
    if (step + 1 < Tsz) {
      const int tn = dir ? (t - 1) : (t + 1);
      const char* g = (const char*)(XP + ((size_t)tn * Bsz + b0) * G);
      char* l = smem + OFF_SLAB0 + ((cur ^ 1) << 15);
#pragma unroll
      for (int i = 0; i < 4; ++i)
        async_copy_b128(l + tid * 16 + i * 8192, g + tid * 16 + i * 8192);
    }

    // A fragments: h (16x128 bf16) in 4 K-chunks, shared across N-tiles.
    bf16x16 af[4];
#pragma unroll
    for (int kc = 0; kc < 4; ++kc) af[kc] = load_a_frag(sH + kc * 32, H, lane);

    // Each wave owns 32 gate columns (2 N-tiles).
    const float* slab = (const float*)(smem + OFF_SLAB0 + (cur << 15));
#pragma unroll
    for (int ni = 0; ni < 2; ++ni) {
      const int n0 = wave * 32 + ni * 16;
      f32x8 acc;
#pragma unroll
      for (int r = 0; r < 8; ++r) acc[r] = slab[(mh + r) * G + n0 + nl];
#pragma unroll
      for (int kc = 0; kc < 4; ++kc) {
        bf16x16 bfrag = load_b_frag(sW + (size_t)n0 * H + kc * 32, H, lane);
        acc = __builtin_amdgcn_wmma_f32_16x16x32_bf16(false, af[kc], false, bfrag,
                                                      (short)0, acc, false, false);
      }
#pragma unroll
      for (int r = 0; r < 8; ++r) sGat[(mh + r) * G + n0 + nl] = acc[r];
    }
    __syncthreads();

    // Cell update: gates order (i, f, g, o), 16 rows x 128 hidden.
    for (int it = tid; it < 2048; it += 512) {
      const int row = it >> 7;
      const int hh  = it & 127;
      const float gi = sGat[row * G + hh];
      const float gf = sGat[row * G + 128 + hh];
      const float gg = sGat[row * G + 256 + hh];
      const float go = sGat[row * G + 384 + hh];
      const float si = 1.0f / (1.0f + __expf(-gi));
      const float sf = 1.0f / (1.0f + __expf(-gf));
      const float so = 1.0f / (1.0f + __expf(-go));
      const float c  = sf * sC[it] + si * tanhf(gg);
      sC[it] = c;
      const float h = so * tanhf(c);
      sH[it] = (bf16_t)h;
      if (Xout)   // layer-1 input, [b][t][dir*H + hh], K-contiguous for the GEMM
        Xout[((size_t)(b0 + row) * Tsz + t) * (2 * H) + dir * H + hh] = (bf16_t)h;
      if (HFinal && step == Tsz - 1)
        HFinal[(size_t)(b0 + row) * (2 * H) + dir * H + hh] = h;
    }

    // Next slab must have landed; also publishes sH/sC for the next step.
    wait_asynccnt0();
    __syncthreads();
    cur ^= 1;
  }
}

// ---------------------------------------------------------------------------
// Host-side orchestration.
// Inputs: 0:x, then per (l0f,l0b,l1f,l1b): w_ih, w_hh, b_ih, b_hh.
// ---------------------------------------------------------------------------
extern "C" void kernel_launch(void* const* d_in, const int* in_sizes, int n_in,
                              void* d_out, int out_size, void* d_ws, size_t ws_size,
                              hipStream_t stream) {
  (void)in_sizes; (void)n_in; (void)out_size; (void)ws_size;

  const float* x = (const float*)d_in[0];
  const float* w_ih[4] = { (const float*)d_in[1], (const float*)d_in[5],
                           (const float*)d_in[9], (const float*)d_in[13] };
  const float* w_hh[4] = { (const float*)d_in[2], (const float*)d_in[6],
                           (const float*)d_in[10], (const float*)d_in[14] };
  const float* b_ih[4] = { (const float*)d_in[3], (const float*)d_in[7],
                           (const float*)d_in[11], (const float*)d_in[15] };
  const float* b_hh[4] = { (const float*)d_in[4], (const float*)d_in[8],
                           (const float*)d_in[12], (const float*)d_in[16] };

  // Workspace layout.
  char* ws = (char*)d_ws;
  size_t off = 0;
  bf16_t* xb  = (bf16_t*)(ws + off); off += (size_t)BT * Din * sizeof(bf16_t);   // 64MB
  bf16_t* x1b = (bf16_t*)(ws + off); off += (size_t)BT * 2 * H * sizeof(bf16_t); // 64MB
  bf16_t* wihb[4];
  for (int i = 0; i < 4; ++i) { wihb[i] = (bf16_t*)(ws + off); off += (size_t)G * Din * sizeof(bf16_t); }
  bf16_t* whhb[4];
  for (int i = 0; i < 4; ++i) { whhb[i] = (bf16_t*)(ws + off); off += (size_t)G * H * sizeof(bf16_t); }
  float* xpF = (float*)(ws + off); off += (size_t)Tsz * Bsz * G * sizeof(float); // 256MB
  float* xpB = (float*)(ws + off); off += (size_t)Tsz * Bsz * G * sizeof(float); // 256MB

  // 1) bf16 conversions.
  {
    int n = BT * Din;
    cvt_f32_bf16<<<dim3((n + 1023) / 1024), dim3(256), 0, stream>>>(x, xb, n);
    for (int i = 0; i < 4; ++i) {
      cvt_f32_bf16<<<dim3((G * Din + 1023) / 1024), dim3(256), 0, stream>>>(w_ih[i], wihb[i], G * Din);
      cvt_f32_bf16<<<dim3((G * H + 1023) / 1024), dim3(256), 0, stream>>>(w_hh[i], whhb[i], G * H);
    }
  }

  const dim3 gGemm(BT / 64, G / 128, 2);

  // 2) Layer 0: input projections (both dirs), then scan.
  xproj_gemm<<<gGemm, dim3(256), 0, stream>>>(xb, wihb[0], wihb[1],
                                              b_ih[0], b_hh[0], b_ih[1], b_hh[1],
                                              xpF, xpB);
  lstm_scan<<<dim3(Bsz / 16, 2), dim3(512), SCAN_SMEM, stream>>>(
      xpF, xpB, whhb[0], whhb[1], x1b, nullptr);

  // 3) Layer 1: reuse xproj scratch (Din == 2H == 256), scan writes d_out.
  xproj_gemm<<<gGemm, dim3(256), 0, stream>>>(x1b, wihb[2], wihb[3],
                                              b_ih[2], b_hh[2], b_ih[3], b_hh[3],
                                              xpF, xpB);
  lstm_scan<<<dim3(Bsz / 16, 2), dim3(512), SCAN_SMEM, stream>>>(
      xpF, xpB, whhb[2], whhb[3], nullptr, (float*)d_out);
}
